// SELayer_56547539419638
// MI455X (gfx1250) — matrix operand order repaired and checked by reading, not compiled, not run on hardware
//
#include <hip/hip_runtime.h>

// ---------------------------------------------------------------------------
// SE layer: x[32,768,56,56] f32
//   s = mean_{h,w}(x); h = relu(s @ w1^T); a = sigmoid(h @ w2^T); out = x * a
// Bandwidth-bound (~925 MB HBM traffic -> ~40us @ 23.3 TB/s). FC GEMMs done
// with v_wmma_f32_16x16x32_f16 using hi/lo f16 splitting for ~fp32 accuracy.
// ---------------------------------------------------------------------------

#define SE_B   32
#define SE_C   768
#define SE_R   48
#define SE_HW  3136          // 56*56
#define SE_HW4 784           // 3136/4

typedef __attribute__((ext_vector_type(16))) _Float16 v16h;
typedef __attribute__((ext_vector_type(8)))  float    v8f;
typedef __attribute__((ext_vector_type(4)))  float    f4;

// ---------------------------------------------------------------------------
// Kernel 1: global average pool. One block per (b,c) channel.
// ---------------------------------------------------------------------------
__global__ __launch_bounds__(256) void se_pool_kernel(const float* __restrict__ x,
                                                      float* __restrict__ s)
{
    const int ch = blockIdx.x;                       // ch = b*SE_C + c
    const f4* px = (const f4*)(x + (size_t)ch * SE_HW);

    float sum = 0.0f;
    for (int i = threadIdx.x; i < SE_HW4; i += 256) {
        f4 v = px[i];
        sum += (v.x + v.y) + (v.z + v.w);
    }
    // wave32 reduction
    #pragma unroll
    for (int off = 16; off > 0; off >>= 1)
        sum += __shfl_down(sum, off, 32);

    __shared__ float red[8];
    if ((threadIdx.x & 31) == 0) red[threadIdx.x >> 5] = sum;
    __syncthreads();
    if (threadIdx.x == 0) {
        float t = 0.0f;
        #pragma unroll
        for (int i = 0; i < 8; ++i) t += red[i];
        s[ch] = t * (1.0f / (float)SE_HW);
    }
}

// ---------------------------------------------------------------------------
// Kernel 2: both FC layers in a single 256-thread (8-wave) block using WMMA.
// fp32 operands are split into hi/lo f16 halves; each K-step does
//   acc += Ahi*Bhi + Ahi*Blo + Alo*Bhi   (f32 accumulate)
// giving ~22 mantissa bits of operand precision (near-fp32 result).
//
// VGPR layouts (ISA 7.12.2, 16x16x32 f16):
//   A (16x32): m = lane%16 ; element e -> k = (e/8)*16 + (e%8) + (lane/16)*8
//   B (32x16): n = lane%16 ; element e -> k as for A
//   C/D (16x16 f32): n = lane%16 ; vgpr v -> m = v + (lane/16)*8
// ---------------------------------------------------------------------------
__device__ __forceinline__ void split_hi_lo(float f, _Float16& hi, _Float16& lo)
{
    hi = (_Float16)f;
    lo = (_Float16)(f - (float)hi);
}

__global__ __launch_bounds__(256) void se_fc_kernel(const float* __restrict__ s,
                                                    const float* __restrict__ w1,
                                                    const float* __restrict__ w2,
                                                    float* __restrict__ a_out)
{
    __shared__ float h_lds[SE_B * SE_R];             // 6 KB

    const int lane = threadIdx.x & 31;
    const int wave = threadIdx.x >> 5;
    const int lh   = lane >> 4;                      // lane half (0/1)
    const int lm   = lane & 15;

    // ---------------- GEMM1: h[32,48] = relu(s[32,768] @ w1[48,768]^T) ------
    if (wave < 6) {
        const int mt = wave / 3;                     // 0..1 (M tiles)
        const int nt = wave % 3;                     // 0..2 (N tiles)
        const int m  = mt * 16 + lm;
        v8f acc = {};

        for (int kt = 0; kt < SE_C; kt += 32) {
            v16h a_hi, a_lo, b_hi, b_lo;
            #pragma unroll
            for (int e = 0; e < 16; ++e) {
                const int k = kt + (e >> 3) * 16 + (e & 7) + lh * 8;
                _Float16 hi, lo;
                split_hi_lo(s[m * SE_C + k], hi, lo);
                a_hi[e] = hi; a_lo[e] = lo;
            }
            const int n = nt * 16 + lm;              // B: lane indexes n
            #pragma unroll
            for (int e = 0; e < 16; ++e) {
                const int kb = kt + (e >> 3) * 16 + (e & 7) + lh * 8;
                _Float16 hi, lo;
                split_hi_lo(w1[n * SE_C + kb], hi, lo);
                b_hi[e] = hi; b_lo[e] = lo;
            }
            acc = __builtin_amdgcn_wmma_f32_16x16x32_f16(false, a_hi, false, b_hi,
                                                         (short)0, acc, false, false);
            acc = __builtin_amdgcn_wmma_f32_16x16x32_f16(false, a_hi, false, b_lo,
                                                         (short)0, acc, false, false);
            acc = __builtin_amdgcn_wmma_f32_16x16x32_f16(false, a_lo, false, b_hi,
                                                         (short)0, acc, false, false);
        }
        #pragma unroll
        for (int v = 0; v < 8; ++v) {
            const int mm = mt * 16 + v + lh * 8;
            const int nn = nt * 16 + lm;
            h_lds[mm * SE_R + nn] = fmaxf(acc[v], 0.0f);   // ReLU
        }
    }
    __syncthreads();

    // ---------------- GEMM2: a[32,768] = sigmoid(h[32,48] @ w2[768,48]^T) ---
    // 2 x 48 = 96 output tiles; 12 per wave. K=48 padded to 64 (zeros).
    for (int t = wave; t < 96; t += 8) {
        const int mt = t / 48;
        const int nt = t % 48;
        const int m  = mt * 16 + lm;
        v8f acc = {};

        #pragma unroll
        for (int kt = 0; kt < 64; kt += 32) {
            v16h a_hi, a_lo, b_hi, b_lo;
            #pragma unroll
            for (int e = 0; e < 16; ++e) {
                const int k = kt + (e >> 3) * 16 + (e & 7) + lh * 8;
                const float f = (k < SE_R) ? h_lds[m * SE_R + k] : 0.0f;
                _Float16 hi, lo;
                split_hi_lo(f, hi, lo);
                a_hi[e] = hi; a_lo[e] = lo;
            }
            const int n = nt * 16 + lm;
            #pragma unroll
            for (int e = 0; e < 16; ++e) {
                const int kb = kt + (e >> 3) * 16 + (e & 7) + lh * 8;
                const float f = (kb < SE_R) ? w2[n * SE_R + kb] : 0.0f;
                _Float16 hi, lo;
                split_hi_lo(f, hi, lo);
                b_hi[e] = hi; b_lo[e] = lo;
            }
            acc = __builtin_amdgcn_wmma_f32_16x16x32_f16(false, a_hi, false, b_hi,
                                                         (short)0, acc, false, false);
            acc = __builtin_amdgcn_wmma_f32_16x16x32_f16(false, a_hi, false, b_lo,
                                                         (short)0, acc, false, false);
            acc = __builtin_amdgcn_wmma_f32_16x16x32_f16(false, a_lo, false, b_hi,
                                                         (short)0, acc, false, false);
        }
        #pragma unroll
        for (int v = 0; v < 8; ++v) {
            const int b = mt * 16 + v + lh * 8;
            const int c = nt * 16 + lm;
            a_out[b * SE_C + c] = 1.0f / (1.0f + __expf(-acc[v]));  // sigmoid
        }
    }
}

// ---------------------------------------------------------------------------
// Kernel 3: out = x * a[b,c]. One block per channel; scale is wave-uniform
// (scalar load). Non-temporal float4 streaming: no reuse, keep L2 clean.
// ---------------------------------------------------------------------------
__global__ __launch_bounds__(256) void se_scale_kernel(const float* __restrict__ x,
                                                       const float* __restrict__ a,
                                                       float* __restrict__ out)
{
    const int ch = blockIdx.x;                       // ch = b*SE_C + c
    const float scale = a[ch];                       // uniform -> s_load
    const f4* px = (const f4*)(x   + (size_t)ch * SE_HW);
    f4*       po = (f4*)      (out + (size_t)ch * SE_HW);

    for (int i = threadIdx.x; i < SE_HW4; i += 256) {
        f4 v = __builtin_nontemporal_load(&px[i]);
        v *= scale;
        __builtin_nontemporal_store(v, &po[i]);
    }
}

// ---------------------------------------------------------------------------
extern "C" void kernel_launch(void* const* d_in, const int* in_sizes, int n_in,
                              void* d_out, int out_size, void* d_ws, size_t ws_size,
                              hipStream_t stream)
{
    (void)in_sizes; (void)n_in; (void)out_size; (void)ws_size;

    const float* x  = (const float*)d_in[0];         // [32,768,56,56]
    const float* w1 = (const float*)d_in[1];         // [48,768]
    const float* w2 = (const float*)d_in[2];         // [768,48]
    float* out = (float*)d_out;                      // [32,768,56,56]

    float* s = (float*)d_ws;                         // [32,768]
    float* a = s + SE_B * SE_C;                      // [32,768]

    const int nch = SE_B * SE_C;                     // 24576 channels

    se_pool_kernel <<<nch, 256, 0, stream>>>(x, s);
    se_fc_kernel   <<<1,    256, 0, stream>>>(s, w1, w2, a);
    se_scale_kernel<<<nch, 256, 0, stream>>>(x, a, out);
}
